// MultiHeadAttentionBlock_3246995275955
// MI455X (gfx1250) — matrix operand order repaired and checked by reading, not compiled, not run on hardware
//
#include <hip/hip_runtime.h>
#include <stdint.h>

// ---------------------------------------------------------------------------
// CDNA5 (gfx1250) MHA block: projections + cosine-attention + output proj,
// all GEMMs via v_wmma_f32_16x16x32_bf16 (wave32).
// ---------------------------------------------------------------------------

typedef __attribute__((ext_vector_type(16))) __bf16 v16bf;
typedef __attribute__((ext_vector_type(8)))  float  v8f;

union BF16x16 { v16bf v; uint4 q[2]; };

#define WMMA_BF16(A, B, C) \
  __builtin_amdgcn_wmma_f32_16x16x32_bf16(false, (A), false, (B), (short)0, (C), false, false)

static __device__ __forceinline__ uint32_t bf16r(float f) {
  uint32_t u = __float_as_uint(f);
  return (u + 0x7FFFu + ((u >> 16) & 1u)) >> 16;  // RNE f32 -> bf16 bits
}
static __device__ __forceinline__ uint32_t pk2bf(float a, float b) {
  return bf16r(a) | (bf16r(b) << 16);
}
static __device__ __forceinline__ float rmax16(float x) {
  x = fmaxf(x, __shfl_xor(x, 1, 32));
  x = fmaxf(x, __shfl_xor(x, 2, 32));
  x = fmaxf(x, __shfl_xor(x, 4, 32));
  x = fmaxf(x, __shfl_xor(x, 8, 32));
  return x;
}
static __device__ __forceinline__ float rsum16(float x) {
  x += __shfl_xor(x, 1, 32);
  x += __shfl_xor(x, 2, 32);
  x += __shfl_xor(x, 4, 32);
  x += __shfl_xor(x, 8, 32);
  return x;
}
static __device__ __forceinline__ float rsum32(float x) {
  x = rsum16(x);
  x += __shfl_xor(x, 16, 32);
  return x;
}

// ---------------------------------------------------------------------------
// GEMM: Y[M=8192,512] = X[8192,512] @ W[512,512]^T   (Y[m,n] = sum_k X[m,k]W[n,k])
// Block = 128x128, K-step 32 (bf16). 8 waves: wave = 64(M) x 32(N) patch.
// LDS rows padded to 112B so fragment chunks are 16B-aligned ds_load_b128.
// ---------------------------------------------------------------------------
#define LDSU 28  // uint32 per LDS row (32 bf16 data + pad = 112 bytes)

__global__ __launch_bounds__(256)
void gemm_xwt_512(const float* __restrict__ X, const float* __restrict__ W,
                  float* __restrict__ Y)
{
  __shared__ __attribute__((aligned(16))) uint32_t lX[128 * LDSU];
  __shared__ __attribute__((aligned(16))) uint32_t lW[128 * LDSU];
  const int t    = threadIdx.x;
  const int lane = t & 31;
  const int wv   = t >> 5;
  const int wm   = wv & 1;    // M half (64 rows)
  const int wn   = wv >> 1;   // N quarter (32 cols)
  const int bm   = blockIdx.y * 128;
  const int bn   = blockIdx.x * 128;
  const int lr   = lane & 15;
  const int kh   = lane >> 4;

  const v8f vz = {0.f,0.f,0.f,0.f,0.f,0.f,0.f,0.f};
  v8f acc[4][2];
#pragma unroll
  for (int i = 0; i < 4; ++i) { acc[i][0] = vz; acc[i][1] = vz; }

  const int lrow = t >> 3;       // 0..31 staging row
  const int c4   = (t & 7) * 4;  // staging col (f32 units)

  for (int kb = 0; kb < 512; kb += 32) {
    __syncthreads();
#pragma unroll
    for (int p = 0; p < 4; ++p) {
      const int row = p * 32 + lrow;
      const float4 xv  = *(const float4*)(X + (size_t)(bm + row) * 512 + kb + c4);
      const float4 wvv = *(const float4*)(W + (size_t)(bn + row) * 512 + kb + c4);
      uint2 xp, wp;
      xp.x = pk2bf(xv.x,  xv.y);  xp.y = pk2bf(xv.z,  xv.w);
      wp.x = pk2bf(wvv.x, wvv.y); wp.y = pk2bf(wvv.z, wvv.w);
      *(uint2*)&lX[row * LDSU + (c4 >> 1)] = xp;
      *(uint2*)&lW[row * LDSU + (c4 >> 1)] = wp;
    }
    __syncthreads();

    BF16x16 af[4], bf[2];
#pragma unroll
    for (int mi = 0; mi < 4; ++mi) {  // A: K pairs {0-7,16-23}/{8-15,24-31} per half
      const uint32_t* p = &lX[(wm * 64 + mi * 16 + lr) * LDSU];
      af[mi].q[0] = *(const uint4*)(p + kh * 4);
      af[mi].q[1] = *(const uint4*)(p + 8 + kh * 4);
    }
#pragma unroll
    for (int ni = 0; ni < 2; ++ni) {  // B: K pairs {0-7,16-23}/{8-15,24-31} per half
      const uint32_t* p = &lW[(wn * 32 + ni * 16 + lr) * LDSU];
      bf[ni].q[0] = *(const uint4*)(p + kh * 4);
      bf[ni].q[1] = *(const uint4*)(p + 8 + kh * 4);
    }
#pragma unroll
    for (int mi = 0; mi < 4; ++mi)
#pragma unroll
      for (int ni = 0; ni < 2; ++ni)
        acc[mi][ni] = WMMA_BF16(af[mi].v, bf[ni].v, acc[mi][ni]);
  }

#pragma unroll
  for (int mi = 0; mi < 4; ++mi)
#pragma unroll
    for (int ni = 0; ni < 2; ++ni)
#pragma unroll
      for (int e = 0; e < 8; ++e) {
        const int row = bm + wm * 64 + mi * 16 + e + 8 * kh;
        const int col = bn + wn * 32 + ni * 16 + lr;
        Y[(size_t)row * 512 + col] = acc[mi][ni][e];
      }
}

// ---------------------------------------------------------------------------
// LayerNorm + L2-normalize Q,K (1/8 score scale folded into Q), emit bf16.
// V: convert to bf16 and transpose into Vt[B,H,DK,S] for contiguous PV frags.
// One wave per row; 8 rows per block.
// ---------------------------------------------------------------------------
__global__ __launch_bounds__(256)
void norm_qkv(const float* __restrict__ Qp, const float* __restrict__ Kp,
              const float* __restrict__ Vp,
              const float* __restrict__ gq, const float* __restrict__ bq,
              const float* __restrict__ gk, const float* __restrict__ bk,
              uint16_t* __restrict__ Qn, uint16_t* __restrict__ Kn,
              uint16_t* __restrict__ Vt)
{
  const int row  = blockIdx.x * 8 + (threadIdx.x >> 5);
  const int lane = threadIdx.x & 31;

#pragma unroll
  for (int which = 0; which < 2; ++which) {
    const float* P = which ? Kp : Qp;
    const float* g = which ? gk : gq;
    const float* b = which ? bk : bq;
    uint16_t* Out  = which ? Kn : Qn;
    const float outscale = which ? 1.0f : 0.125f;  // 1/sqrt(d_k) into Q

    float x[16];
    float s = 0.f, s2 = 0.f;
#pragma unroll
    for (int i = 0; i < 16; ++i) {
      x[i] = P[(size_t)row * 512 + lane + i * 32];
      s += x[i]; s2 += x[i] * x[i];
    }
    s = rsum32(s); s2 = rsum32(s2);
    const float mean = s * (1.f / 512.f);
    const float var  = s2 * (1.f / 512.f) - mean * mean;
    const float rstd = rsqrtf(var + 1e-5f);

    float y[16]; float n2 = 0.f;
#pragma unroll
    for (int i = 0; i < 16; ++i) {
      y[i] = (x[i] - mean) * rstd * g[lane + i * 32] + b[lane + i * 32];
      n2 += y[i] * y[i];
    }
    n2 = rsum32(n2);
    const float sc = outscale / (sqrtf(n2) + 1e-6f);
#pragma unroll
    for (int i = 0; i < 16; ++i)
      Out[(size_t)row * 512 + lane + i * 32] = (uint16_t)bf16r(y[i] * sc);
  }

  const int bb = row >> 11;
  const int ss = row & 2047;
  for (int i = lane; i < 512; i += 32) {
    const float vv = Vp[(size_t)row * 512 + i];
    const int hh = i >> 6, dd = i & 63;
    Vt[((size_t)(bb * 8 + hh) * 64 + dd) * 2048 + ss] = (uint16_t)bf16r(vv);
  }
}

// ---------------------------------------------------------------------------
// Flash attention per (b,h): wave = 16 query rows, streams 32-key blocks.
// scores = Q@K^T (pre-scaled), mask, online softmax, O += P@V.
// P transposed C-layout -> A-layout through a private per-wave LDS tile.
// ---------------------------------------------------------------------------
__global__ __launch_bounds__(128)
void attn_flash(const uint16_t* __restrict__ Qn, const uint16_t* __restrict__ Kn,
                const uint16_t* __restrict__ Vt, const int* __restrict__ mask,
                float* __restrict__ O)
{
  __shared__ __attribute__((aligned(16))) uint16_t lP[4][16 * 56];  // 112B rows
  const int t    = threadIdx.x;
  const int lane = t & 31;
  const int wv   = t >> 5;
  const int lr   = lane & 15;
  const int kh   = lane >> 4;
  const int hh   = blockIdx.y;
  const int bb   = blockIdx.z;
  const int qrow = bb * 2048 + blockIdx.x * 64 + wv * 16;

  BF16x16 qa[2];
  {
    const uint16_t* qp = Qn + (size_t)(qrow + lr) * 512 + hh * 64;
#pragma unroll
    for (int d = 0; d < 2; ++d) {
      qa[d].q[0] = *(const uint4*)(qp + d * 32 + kh * 16);
      qa[d].q[1] = *(const uint4*)(qp + d * 32 + kh * 16 + 8);
    }
  }

  const v8f vz = {0.f,0.f,0.f,0.f,0.f,0.f,0.f,0.f};
  v8f o0 = vz, o1 = vz, o2 = vz, o3 = vz;
  float m[8], l[8];
#pragma unroll
  for (int i = 0; i < 8; ++i) { m[i] = -1e30f; l[i] = 0.f; }

  const int*      mrow = mask + bb * 2048;
  const uint16_t* kbp  = Kn + (size_t)(bb * 2048) * 512 + hh * 64;
  const uint16_t* vbp  = Vt + (size_t)(bb * 8 + hh) * 64 * 2048;
  uint16_t*       myP  = &lP[wv][0];

  for (int kb = 0; kb < 2048; kb += 32) {
    v8f s0 = vz, s1 = vz;
    {
      const uint16_t* kp = kbp + (size_t)(kb + lr) * 512;   // keys kb..kb+15
      BF16x16 f0, f1;
      f0.q[0] = *(const uint4*)(kp + kh * 16);
      f0.q[1] = *(const uint4*)(kp + kh * 16 + 8);
      f1.q[0] = *(const uint4*)(kp + 32 + kh * 16);
      f1.q[1] = *(const uint4*)(kp + 32 + kh * 16 + 8);
      s0 = WMMA_BF16(qa[0].v, f0.v, s0);
      s0 = WMMA_BF16(qa[1].v, f1.v, s0);
      const uint16_t* kp2 = kp + (size_t)16 * 512;          // keys kb+16..kb+31
      f0.q[0] = *(const uint4*)(kp2 + kh * 16);
      f0.q[1] = *(const uint4*)(kp2 + kh * 16 + 8);
      f1.q[0] = *(const uint4*)(kp2 + 32 + kh * 16);
      f1.q[1] = *(const uint4*)(kp2 + 32 + kh * 16 + 8);
      s1 = WMMA_BF16(qa[0].v, f0.v, s1);
      s1 = WMMA_BF16(qa[1].v, f1.v, s1);
    }

    const int mk0 = mrow[kb + lr];
    const int mk1 = mrow[kb + 16 + lr];
    float al[8];
#pragma unroll
    for (int e = 0; e < 8; ++e) {
      float a0 = mk0 ? s0[e] : -10000.f;
      float a1 = mk1 ? s1[e] : -10000.f;
      float nm = fmaxf(m[e], rmax16(fmaxf(a0, a1)));
      float alpha = __expf(m[e] - nm);
      float p0 = __expf(a0 - nm);
      float p1 = __expf(a1 - nm);
      l[e] = l[e] * alpha + rsum16(p0 + p1);
      m[e] = nm; al[e] = alpha;
      myP[(e + 8 * kh) * 56 + lr]      = (uint16_t)bf16r(p0);
      myP[(e + 8 * kh) * 56 + 16 + lr] = (uint16_t)bf16r(p1);
    }
#pragma unroll
    for (int e = 0; e < 8; ++e) {
      o0[e] *= al[e]; o1[e] *= al[e]; o2[e] *= al[e]; o3[e] *= al[e];
    }

    asm volatile("s_wait_dscnt 0" ::: "memory");  // wave-private LDS RAW

    BF16x16 pa;
    {
      const uint16_t* pr = myP + lr * 56;
      pa.q[0] = *(const uint4*)(pr + kh * 16);
      pa.q[1] = *(const uint4*)(pr + kh * 16 + 8);
    }
    {
      BF16x16 vf;
      const uint16_t* vp = vbp + (size_t)lr * 2048 + kb;
      vf.q[0] = *(const uint4*)(vp + kh * 16);
      vf.q[1] = *(const uint4*)(vp + kh * 16 + 8);
      o0 = WMMA_BF16(pa.v, vf.v, o0);
      vp += (size_t)16 * 2048;
      vf.q[0] = *(const uint4*)(vp + kh * 16);
      vf.q[1] = *(const uint4*)(vp + kh * 16 + 8);
      o1 = WMMA_BF16(pa.v, vf.v, o1);
      vp += (size_t)16 * 2048;
      vf.q[0] = *(const uint4*)(vp + kh * 16);
      vf.q[1] = *(const uint4*)(vp + kh * 16 + 8);
      o2 = WMMA_BF16(pa.v, vf.v, o2);
      vp += (size_t)16 * 2048;
      vf.q[0] = *(const uint4*)(vp + kh * 16);
      vf.q[1] = *(const uint4*)(vp + kh * 16 + 8);
      o3 = WMMA_BF16(pa.v, vf.v, o3);
    }
  }

#pragma unroll
  for (int e = 0; e < 8; ++e) l[e] = 1.f / l[e];
#pragma unroll
  for (int e = 0; e < 8; ++e) {
    float* orow = O + (size_t)(qrow + e + 8 * kh) * 512 + hh * 64 + lr;
    orow[0]  = o0[e] * l[e];
    orow[16] = o1[e] * l[e];
    orow[32] = o2[e] * l[e];
    orow[48] = o3[e] * l[e];
  }
}

// ---------------------------------------------------------------------------
// Host-side launch
// ---------------------------------------------------------------------------
extern "C" void kernel_launch(void* const* d_in, const int* in_sizes, int n_in,
                              void* d_out, int out_size, void* d_ws, size_t ws_size,
                              hipStream_t stream)
{
  (void)in_sizes; (void)n_in; (void)out_size; (void)ws_size;
  const float* q    = (const float*)d_in[0];
  const float* k    = (const float*)d_in[1];
  const float* v    = (const float*)d_in[2];
  const int*   mask = (const int*)  d_in[3];
  const float* Wq   = (const float*)d_in[4];
  const float* Wk   = (const float*)d_in[5];
  const float* Wv   = (const float*)d_in[6];
  const float* Wo   = (const float*)d_in[7];
  const float* gq   = (const float*)d_in[8];
  const float* bq   = (const float*)d_in[9];
  const float* gk   = (const float*)d_in[10];
  const float* bk   = (const float*)d_in[11];
  float* out = (float*)d_out;

  const size_t NE = (size_t)8192 * 512;
  float* fws = (float*)d_ws;
  float* Qp = fws;            // 16 MB
  float* Kp = fws + NE;       // 16 MB
  float* Vp = fws + 2 * NE;   // 16 MB
  uint16_t* u16 = (uint16_t*)(fws + 3 * NE);
  uint16_t* Qn = u16;             // 8 MB bf16
  uint16_t* Kn = u16 + NE;        // 8 MB bf16
  uint16_t* Vt = u16 + 2 * NE;    // 8 MB bf16, [B,H,DK,S]
  float* Oattn = fws;             // reuse Qp region (consumed before attn)

  const dim3 gg(4, 64), gb(256);
  gemm_xwt_512<<<gg, gb, 0, stream>>>(q, Wq, Qp);
  gemm_xwt_512<<<gg, gb, 0, stream>>>(k, Wk, Kp);
  gemm_xwt_512<<<gg, gb, 0, stream>>>(v, Wv, Vp);
  norm_qkv<<<1024, 256, 0, stream>>>(Qp, Kp, Vp, gq, bq, gk, bk, Qn, Kn, Vt);
  attn_flash<<<dim3(32, 8, 4), 128, 0, stream>>>(Qn, Kn, Vt, mask, Oattn);
  gemm_xwt_512<<<gg, gb, 0, stream>>>(Oattn, Wo, out);
}
